// MelFreeStemEmbedMamba_12146167513468
// MI455X (gfx1250) — compile-verified, matrix operand order checked by reading, not run on hardware
//
#include <hip/hip_runtime.h>
#include <hip/hip_bf16.h>

// ---------------------------------------------------------------------------
// MelFreeStemEmbedMamba for MI455X (gfx1250, wave32, WMMA)
//
// Token-major (B*L, C) layout everywhere. All four Mamba projections run on
// v_wmma_f32_16x16x32_f16 (f16 operands, f32 accumulate). The selective scan
// is fully fused (dA/dBx/h-update/C-reduction/D-skip/silu(z) gate) so the
// (B,L,di,16) tensors are never materialized. Convs / LN / elementwise are
// small and latency-bound -> simple VALU kernels.
// ---------------------------------------------------------------------------

typedef _Float16 half8  __attribute__((ext_vector_type(8)));
typedef _Float16 half16 __attribute__((ext_vector_type(16)));
typedef float    float8 __attribute__((ext_vector_type(8)));

static inline int cdiv_i(int a, int b) { return (a + b - 1) / b; }

// ---------------- elementwise ----------------

__global__ void k_copy(const float* __restrict__ s, float* __restrict__ d, int n) {
  int i = blockIdx.x * blockDim.x + threadIdx.x;
  if (i < n) d[i] = s[i];
}

__global__ void k_leaky(const float* __restrict__ s, float* __restrict__ d, int n) {
  int i = blockIdx.x * blockDim.x + threadIdx.x;
  if (i < n) { float v = s[i]; d[i] = v >= 0.f ? v : 0.2f * v; }
}

__global__ void k_softplus(float* __restrict__ x, int n) {
  int i = blockIdx.x * blockDim.x + threadIdx.x;
  if (i < n) { float v = x[i]; x[i] = v > 20.f ? v : log1pf(__expf(v)); }
}

// f32 (rows x cols, leading dim ldSrc) -> f16 (rowsPad x colsPad), zero pad.
__global__ void k_cvt_f16(const float* __restrict__ src, int ldSrc,
                          _Float16* __restrict__ dst,
                          int rows, int rowsPad, int cols, int colsPad) {
  int i = blockIdx.x * blockDim.x + threadIdx.x;
  int total = rowsPad * colsPad;
  if (i >= total) return;
  int r = i / colsPad, c = i % colsPad;
  float v = (r < rows && c < cols) ? src[(size_t)r * ldSrc + c] : 0.f;
  dst[i] = (_Float16)v;
}

// (B,L,C) token-major -> (B,C,L) reference layout
__global__ void k_tr_blc_bcl(const float* __restrict__ s, float* __restrict__ d,
                             int Bn, int L, int C) {
  int i = blockIdx.x * blockDim.x + threadIdx.x;
  int total = Bn * L * C;
  if (i >= total) return;
  int c = i % C; int t = i / C; int l = t % L; int bb = t / L;
  d[((size_t)bb * C + c) * L + l] = s[i];
}

// ---------------- conv1d (token-major in/out) ----------------

__global__ void k_conv1d(const float* __restrict__ x, const float* __restrict__ w,
                         const float* __restrict__ bias, float* __restrict__ y,
                         int Bn, int Lin, int Cin, int Lout, int Cout,
                         int K, int stride, int pad) {
  int i = blockIdx.x * blockDim.x + threadIdx.x;
  int total = Bn * Lout * Cout;
  if (i >= total) return;
  int co = i % Cout; int t = i / Cout; int lo = t % Lout; int bb = t / Lout;
  float acc = bias[co];
  int base = lo * stride - pad;
  for (int k = 0; k < K; ++k) {
    int li = base + k;
    if (li < 0 || li >= Lin) continue;
    const float* xp = x + (size_t)(bb * Lin + li) * Cin;
    const float* wp = w + (size_t)co * Cin * K + k;
    for (int ci = 0; ci < Cin; ++ci) acc += xp[ci] * wp[(size_t)ci * K];
  }
  y[i] = acc;
}

// ---------------- LayerNorm over channels (one wave32 per token row) --------
// y = LN(x [+ addB]) * g + b [+ addA]

__global__ void __launch_bounds__(256)
k_ln(const float* __restrict__ x, const float* __restrict__ addB,
     const float* __restrict__ addA, const float* __restrict__ g,
     const float* __restrict__ b, float* __restrict__ y, int T, int C) {
  int wave = threadIdx.x >> 5, lane = threadIdx.x & 31;
  int row = blockIdx.x * 8 + wave;
  if (row >= T) return;
  const float* xr = x + (size_t)row * C;
  const float* ar = addB ? addB + (size_t)row * C : nullptr;
  float s = 0.f, s2 = 0.f;
  for (int c = lane; c < C; c += 32) {
    float v = xr[c]; if (ar) v += ar[c];
    s += v; s2 += v * v;
  }
  for (int o = 16; o > 0; o >>= 1) {
    s  += __shfl_xor(s,  o, 32);
    s2 += __shfl_xor(s2, o, 32);
  }
  float mean = s / C;
  float var  = s2 / C - mean * mean;
  float rs   = rsqrtf(var + 1e-5f);
  float* yr = y + (size_t)row * C;
  const float* aa = addA ? addA + (size_t)row * C : nullptr;
  for (int c = lane; c < C; c += 32) {
    float v = xr[c]; if (ar) v += ar[c];
    float o2 = (v - mean) * rs * g[c] + b[c];
    if (aa) o2 += aa[c];
    yr[c] = o2;
  }
}

// ---------------- depthwise causal conv (K=4) + SiLU ----------------
// xi[b,l,ch] = silu(bias[ch] + sum_k xz[b, l-3+k, ch] * w[ch,k]); xz ld = 2*di

__global__ void k_dwconv_silu(const float* __restrict__ xz, const float* __restrict__ w,
                              const float* __restrict__ bias, float* __restrict__ xi,
                              int Bn, int L, int di) {
  int i = blockIdx.x * blockDim.x + threadIdx.x;
  int total = Bn * L * di;
  if (i >= total) return;
  int ch = i % di; int t = i / di; int l = t % L; int bb = t / L;
  float acc = bias[ch];
#pragma unroll
  for (int k = 0; k < 4; ++k) {
    int li = l - 3 + k;
    if (li >= 0) acc += xz[(size_t)(bb * L + li) * (2 * di) + ch] * w[ch * 4 + k];
  }
  xi[i] = acc / (1.f + __expf(-acc));
}

// ---------------- fused selective scan ----------------
// One thread per (b, ch): h[16] in registers; y = (sum_n h*C + D*xi) * silu(z)

__global__ void k_scan(const float* __restrict__ dt, const float* __restrict__ xi,
                       const float* __restrict__ dbl, int ldDbl, int dtRank,
                       const float* __restrict__ Alog, const float* __restrict__ Dp,
                       const float* __restrict__ xz, float* __restrict__ y,
                       int Bn, int L, int di) {
  int i = blockIdx.x * blockDim.x + threadIdx.x;
  if (i >= Bn * di) return;
  int ch = i % di, bb = i / di;
  float a[16], h[16];
#pragma unroll
  for (int n = 0; n < 16; ++n) {
    a[n] = -__expf(Alog[ch * 16 + n]);
    h[n] = 0.f;
  }
  float dv = Dp[ch];
  for (int l = 0; l < L; ++l) {
    size_t t = (size_t)bb * L + l;
    float dtv = dt[t * di + ch];
    float xiv = xi[t * di + ch];
    const float* Brow = dbl + t * ldDbl + dtRank;
    const float* Crow = Brow + 16;
    float acc = 0.f;
#pragma unroll
    for (int n = 0; n < 16; ++n) {
      float dA = __expf(dtv * a[n]);
      h[n] = dA * h[n] + dtv * Brow[n] * xiv;
      acc += h[n] * Crow[n];
    }
    float zv = xz[t * (2 * di) + di + ch];
    y[t * di + ch] = (acc + dv * xiv) * (zv / (1.f + __expf(-zv)));
  }
}

// ---------------- WMMA GEMM: C[M,N] = A[M,K] @ W[N,K]^T (+bias) -------------
// A: f16 (M x lda), W: f16 (rowsPad>=N x ldb), K multiple of 32, M mult of 16.
// One 16x16 tile per wave; per-lane fragment loads follow the CDNA5 16-bit
// A/B layout: lanes 0-15 hold K = kb+0..7 & kb+16..23, lanes 16-31 hold
// K = kb+8..15 & kb+24..31 -> two contiguous half8 (16B) loads per fragment.

__global__ void __launch_bounds__(256)
k_gemm_wmma(const _Float16* __restrict__ A, const _Float16* __restrict__ W,
            float* __restrict__ C, const float* __restrict__ bias,
            int M, int N, int Kpad, int lda, int ldb, int ldc) {
  int lane = threadIdx.x & 31;
  int wave = threadIdx.x >> 5;
  int tilesN = (N + 15) >> 4;
  int tilesM = M >> 4;
  int tile = blockIdx.x * 8 + wave;
  if (tile >= tilesM * tilesN) return;
  int tm = tile / tilesN, tn = tile % tilesN;
  int g = lane >> 4, r16 = lane & 15;

  const _Float16* arow = A + (size_t)(tm * 16 + r16) * lda;
  const _Float16* brow = W + (size_t)(tn * 16 + r16) * ldb;

  float8 acc = {};
  for (int kb = 0; kb < Kpad; kb += 32) {
    half8 alo = *(const half8*)(arow + kb + 8 * g);
    half8 ahi = *(const half8*)(arow + kb + 16 + 8 * g);
    half8 blo = *(const half8*)(brow + kb + 8 * g);
    half8 bhi = *(const half8*)(brow + kb + 16 + 8 * g);
    half16 af, bf;
#pragma unroll
    for (int j = 0; j < 8; ++j) {
      af[j] = alo[j]; af[j + 8] = ahi[j];
      bf[j] = blo[j]; bf[j + 8] = bhi[j];
    }
    acc = __builtin_amdgcn_wmma_f32_16x16x32_f16(
        false, af, false, bf, (short)0, acc, false, false);
  }
  int n = tn * 16 + r16;
  if (n < N) {
    float bv = bias ? bias[n] : 0.f;
#pragma unroll
    for (int r = 0; r < 8; ++r) {
      int m = tm * 16 + g * 8 + r;   // C/D layout: lanes 16-31 hold M = r+8
      C[(size_t)m * ldc + n] = acc[r] + bv;
    }
  }
}

// ---------------------------------------------------------------------------
// Host orchestration
// ---------------------------------------------------------------------------

struct MambaP {
  const float *in_w, *conv_w, *conv_b, *x_proj_w, *dt_w, *dt_b, *A_log, *D, *out_w;
};

static void run_mamba(hipStream_t st, const MambaP& p, const float* x, float* out,
                      int Bn, int L, int d,
                      float* xz, float* xi, float* dbl, float* dtb, float* yb,
                      _Float16* AH, _Float16* WH) {
  const int T = Bn * L, di = 2 * d, dtRank = d / 16;
  const int dpad = ((d + 31) / 32) * 32;
  const int nx = dtRank + 32;     // 35 / 38 / 44
  const int NXPAD = 48;

  auto CV = [&](const float* src, int ldSrc, _Float16* dst,
                int rows, int rowsPad, int cols, int colsPad) {
    int tot = rowsPad * colsPad;
    k_cvt_f16<<<cdiv_i(tot, 256), 256, 0, st>>>(src, ldSrc, dst, rows, rowsPad, cols, colsPad);
  };
  auto GEMM = [&](const _Float16* Af, const _Float16* Wf, float* Cf, const float* bias,
                  int M, int N, int Kp, int lda, int ldb, int ldc) {
    int tiles = (M / 16) * ((N + 15) / 16);
    k_gemm_wmma<<<cdiv_i(tiles, 8), 256, 0, st>>>(Af, Wf, Cf, bias, M, N, Kp, lda, ldb, ldc);
  };

  // 1) xz = x @ in_w^T               (T x 2di)
  CV(x, d, AH, T, T, d, dpad);
  CV(p.in_w, d, WH, 2 * di, 2 * di, d, dpad);
  GEMM(AH, WH, xz, nullptr, T, 2 * di, dpad, dpad, dpad, 2 * di);

  // 2) depthwise causal conv + SiLU  -> xi (T x di)
  k_dwconv_silu<<<cdiv_i(T * di, 256), 256, 0, st>>>(xz, p.conv_w, p.conv_b, xi, Bn, L, di);

  // 3) dbl = xi @ x_proj^T           (T x nx, ld NXPAD)
  CV(xi, di, AH, T, T, di, di);
  CV(p.x_proj_w, di, WH, nx, NXPAD, di, di);
  GEMM(AH, WH, dbl, nullptr, T, nx, di, di, di, NXPAD);

  // 4) dt = softplus(dbl[:, :dtRank] @ dt_w^T + dt_b)   (T x di)
  CV(dbl, NXPAD, AH, T, T, dtRank, 32);
  CV(p.dt_w, dtRank, WH, di, di, dtRank, 32);
  GEMM(AH, WH, dtb, p.dt_b, T, di, 32, 32, 32, di);
  k_softplus<<<cdiv_i(T * di, 256), 256, 0, st>>>(dtb, T * di);

  // 5) fused selective scan (+ D skip + silu(z) gate)   -> yb (T x di)
  k_scan<<<cdiv_i(Bn * di, 256), 256, 0, st>>>(dtb, xi, dbl, NXPAD, dtRank,
                                               p.A_log, p.D, xz, yb, Bn, L, di);

  // 6) out = y @ out_w^T             (T x d)
  CV(yb, di, AH, T, T, di, di);
  CV(p.out_w, di, WH, d, d, di, di);
  GEMM(AH, WH, out, nullptr, T, d, di, di, di, d);
}

extern "C" void kernel_launch(void* const* d_in, const int* in_sizes, int n_in,
                              void* d_out, int out_size, void* d_ws, size_t ws_size,
                              hipStream_t stream) {
  (void)in_sizes; (void)n_in; (void)out_size;
  auto P = [&](int i) { return (const float*)d_in[i]; };

  // --- workspace layout (f32 then f16); ~54 MB total ---
  const size_t ACT = 1572864;  // 4*16384*24 max token-major activation
  if (ws_size < (size_t)56 << 20) return;
  float* F  = (float*)d_ws;
  size_t off = 0;
  float* A0  = F + off; off += ACT;
  float* A1  = F + off; off += ACT;
  float* A2  = F + off; off += ACT;
  float* xz  = F + off; off += 3145728;  // T*2di max
  float* xi  = F + off; off += ACT;
  float* dbl = F + off; off += 786432;   // T*48 max
  float* dtb = F + off; off += ACT;
  float* yb  = F + off; off += ACT;
  _Float16* AH = (_Float16*)(F + off);   // T*Kpad max (1,572,864 halves)
  _Float16* WH = AH + ACT;               // weight staging (<=147,456 halves)

  float* out = (float*)d_out;
  const int Bn = 4;

  // ---- d_in indexing: setup_inputs() dict insertion order ----
  // 0:x | stem: 1:w 2:b 3:ln_g 4:ln_b
  // block i (base 5+26*i): +0 w +1 b +2 convln_g +3 convln_b +4 ln1_g +5 ln1_b
  //   +6 ln2_g +7 ln2_b | m1 (+8..+16): in_w conv_w conv_b x_proj_w dt_w dt_b
  //   A_log D out_w | m2 (+17..+25) | final: 83:w 84:b
  auto mk_mamba = [&](int base) {
    MambaP m;
    m.in_w = P(base + 0); m.conv_w = P(base + 1); m.conv_b = P(base + 2);
    m.x_proj_w = P(base + 3); m.dt_w = P(base + 4); m.dt_b = P(base + 5);
    m.A_log = P(base + 6); m.D = P(base + 7); m.out_w = P(base + 8);
    return m;
  };

  // d_out: [final (4,384,512)] [res0 = input (4,1,65536)] [res1 stem (4,24,16384)]
  //        [res2 (4,48,4096)] [res3 (4,96,2048)] [res4 (4,192,1024)]
  const size_t OFF_FINAL = 0, OFF_RES0 = 786432, OFF_RES1 = 1048576;
  const size_t OFF_RESB[3] = {2621440, 3407872, 4194304};

  // res0: raw input copy
  k_copy<<<cdiv_i(262144, 256), 256, 0, stream>>>(P(0), out + OFF_RES0, 262144);

  // ---- stem: conv1d(1->24, K=20, s=4, p=8) -> LN -> (store) -> leaky ----
  {
    int Lin = 65536, Lout = 16384, Cout = 24;
    k_conv1d<<<cdiv_i(Bn * Lout * Cout, 256), 256, 0, stream>>>(
        P(0), P(1), P(2), A0, Bn, Lin, 1, Lout, Cout, 20, 4, 8);
    int T = Bn * Lout;
    k_ln<<<cdiv_i(T, 8), 256, 0, stream>>>(A0, nullptr, nullptr, P(3), P(4), A1, T, Cout);
    k_tr_blc_bcl<<<cdiv_i(T * Cout, 256), 256, 0, stream>>>(A1, out + OFF_RES1, Bn, Lout, Cout);
    k_leaky<<<cdiv_i(T * Cout, 256), 256, 0, stream>>>(A1, A0, T * Cout);
  }

  // ---- blocks ----
  const int LinA[3]  = {16384, 4096, 2048};
  const int CinA[3]  = {24, 48, 96};
  const int CoutA[3] = {48, 96, 192};
  const int KA[3]    = {16, 8, 8};
  const int SA[3]    = {4, 2, 2};

  for (int i = 0; i < 3; ++i) {
    int base = 5 + 26 * i;
    int Lin = LinA[i], Cin = CinA[i], C = CoutA[i];
    int K = KA[i], S = SA[i], pad = (K - S) / 2;
    int Lout = Lin / S, T = Bn * Lout;
    MambaP m1 = mk_mamba(base + 8), m2 = mk_mamba(base + 17);

    // conv -> A1 (res1)
    k_conv1d<<<cdiv_i(T * C, 256), 256, 0, stream>>>(
        A0, P(base + 0), P(base + 1), A1, Bn, Lin, Cin, Lout, C, K, S, pad);
    // x1 = LN(conv) + conv -> A2
    k_ln<<<cdiv_i(T, 8), 256, 0, stream>>>(A1, nullptr, A1, P(base + 2), P(base + 3), A2, T, C);
    // x2 = mamba1(x1) -> A1 (res2)
    run_mamba(stream, m1, A2, A1, Bn, Lout, C, xz, xi, dbl, dtb, yb, AH, WH);
    // x3 = LN(x2) -> A0
    k_ln<<<cdiv_i(T, 8), 256, 0, stream>>>(A1, nullptr, nullptr, P(base + 4), P(base + 5), A0, T, C);
    // x4 = mamba2(x3) -> A2
    run_mamba(stream, m2, A0, A2, Bn, Lout, C, xz, xi, dbl, dtb, yb, AH, WH);
    // x5 = LN(x2 + x4) -> A0
    k_ln<<<cdiv_i(T, 8), 256, 0, stream>>>(A2, A1, nullptr, P(base + 6), P(base + 7), A0, T, C);
    // store res, then leaky in place for next stage input
    k_tr_blc_bcl<<<cdiv_i(T * C, 256), 256, 0, stream>>>(A0, out + OFF_RESB[i], Bn, Lout, C);
    k_leaky<<<cdiv_i(T * C, 256), 256, 0, stream>>>(A0, A0, T * C);
  }

  // ---- final conv: 192 -> 384, K=8, s=2, p=3 ----
  {
    int Lin = 1024, Lout = 512, Cin = 192, Cout = 384;
    k_conv1d<<<cdiv_i(Bn * Lout * Cout, 256), 256, 0, stream>>>(
        A0, P(83), P(84), A1, Bn, Lin, Cin, Lout, Cout, 8, 2, 3);
    k_tr_blc_bcl<<<cdiv_i(Bn * Lout * Cout, 256), 256, 0, stream>>>(
        A1, out + OFF_FINAL, Bn, Lout, Cout);
  }
}
